// Model_52999896432646
// MI455X (gfx1250) — matrix operand order repaired and checked by reading, not compiled, not run on hardware
//
#include <hip/hip_runtime.h>
#include <hip/hip_bf16.h>

// ---------------- model constants ----------------
#define BATCH   2048
#define NSP     8
#define NDN     8
#define HLEN    50
#define ED      16
#define FD      512
#define VOC     100000
#define HID1    256
#define HID2    128
#define NEXP    8
#define NTASK   4
#define EOUT    8
#define IN_DIM  1160
#define IN_PAD  1184          // 1160 padded to multiple of 32 for WMMA K
// dnn_in layout: [0,128) sparse | [128,136) dense | [136,648) feed | [648,1160) user_interest | pad

typedef __attribute__((ext_vector_type(8)))  float  v8f;
typedef __attribute__((ext_vector_type(16))) __bf16 v16bf;

union Frag { unsigned int u[8]; uint4 q[2]; v16bf v; };

// ---------------- bf16 helpers ----------------
static __device__ __forceinline__ unsigned short f2bf(float f) {
    unsigned int u = __float_as_uint(f);
    u += 0x7FFFu + ((u >> 16) & 1u);          // round-to-nearest-even
    return (unsigned short)(u >> 16);
}
static __device__ __forceinline__ float bf2f(unsigned short h) {
    return __uint_as_float(((unsigned int)h) << 16);
}
static __device__ __forceinline__ unsigned int pack2(float a, float b) {
    return (unsigned int)f2bf(a) | ((unsigned int)f2bf(b) << 16);
}
static __device__ __forceinline__ float lo16f(unsigned int u) { return __uint_as_float(u << 16); }
static __device__ __forceinline__ float hi16f(unsigned int u) { return __uint_as_float(u & 0xFFFF0000u); }

static __device__ __forceinline__ v8f wmma_bf16(const Frag& a, const Frag& b, v8f c) {
    // D = A(16x32 bf16) x B(32x16 bf16) + C(16x16 f32)
    return __builtin_amdgcn_wmma_f32_16x16x32_bf16(false, a.v, false, b.v,
                                                   (short)0, c, false, false);
}

// A-fragment (16-bit 16x32): lane holds M=lane%16; u[v] packs K pair
// k = (v>=4?16:0) + (lane>=16?8:0) + (v&3)*2  ->  u[0..3] = 8 consecutive bf16,
// u[4..7] = the 8 bf16 starting 16 further on. Two 16B loads.
static __device__ __forceinline__ void load_frag16(Frag& f, const unsigned short* p) {
    f.q[0] = *(const uint4*)p;
    f.q[1] = *(const uint4*)(p + 16);
}
static __device__ __forceinline__ void load_bfrag(Frag& f, const unsigned int* w) {
    const uint4* bp = (const uint4*)w;
    f.q[0] = bp[0];
    f.q[1] = bp[1];
}

// ---------------- weight pre-pack into WMMA B-fragment layout ----------------
// dst layout: [N/16][Kpad/32][lane(32)][vgpr(8)] of u32 (2 packed bf16 = K pair)
// B fragment mapping: n = nt*16 + (lane&15); k = kc*32 + (v>=4 ? 16 : 0) + (lane>=16 ? 8 : 0) + (v&3)*2
__global__ void pack_b_kernel(const float* __restrict__ src, unsigned int* __restrict__ dst,
                              int Kpad, int N, int Ksrc) {
    int  kch    = Kpad >> 5;
    int  ntiles = N >> 4;
    long total  = (long)ntiles * kch * 256;
    for (long idx = (long)blockIdx.x * blockDim.x + threadIdx.x; idx < total;
         idx += (long)gridDim.x * blockDim.x) {
        int  v    = (int)(idx & 7);
        int  lane = (int)((idx >> 3) & 31);
        long rest = idx >> 8;
        int  kc   = (int)(rest % kch);
        int  nt   = (int)(rest / kch);
        int  n    = nt * 16 + (lane & 15);
        int  k    = kc * 32 + ((v >= 4) ? 16 : 0) + ((lane >= 16) ? 8 : 0) + (v & 3) * 2;
        float a = (k     < Ksrc) ? src[(long)k       * N + n] : 0.f;
        float b = (k + 1 < Ksrc) ? src[(long)(k + 1) * N + n] : 0.f;
        dst[idx] = pack2(a, b);
    }
}

// fuse expert_w [NE][HID2][EO] and gate_w [NTASK][HID2][NE] into one [HID2][96] f32
__global__ void mmoe_stage_kernel(const float* __restrict__ ew, const float* __restrict__ gw,
                                  float* __restrict__ dst) {
    int idx = blockIdx.x * blockDim.x + threadIdx.x;
    if (idx >= HID2 * 96) return;
    int k = idx / 96, n = idx % 96;
    float v;
    if (n < 64) { int e = n >> 3, o = n & 7; v = ew[((size_t)e * HID2 + k) * EOUT + o]; }
    else        { int nn = n - 64; int t = nn >> 3, e = nn & 7;
                  v = gw[((size_t)t * HID2 + k) * NEXP + e]; }
    dst[idx] = v;
}

// ---------------- gather: sparse emb + dense + feed -> bf16 dnn_in staging ----------------
__global__ __launch_bounds__(256) void gather_kernel(
        const int* __restrict__ sparse_ids, const float* __restrict__ dense,
        const int* __restrict__ feedid, const float* __restrict__ emb_tables,
        const float* __restrict__ feed_table, unsigned short* __restrict__ dnn_in) {
    int b = blockIdx.x, tid = threadIdx.x;
    unsigned short* row = dnn_in + (size_t)b * IN_PAD;
    if (tid < 128) {
        int s = tid >> 4, d = tid & 15;
        int id = sparse_ids[b * NSP + s];
        row[tid] = f2bf(emb_tables[((size_t)s * VOC + id) * ED + d]);
    } else if (tid < 136) {
        row[tid] = f2bf(dense[b * NDN + (tid - 128)]);
    } else if (tid < 160) {
        row[1024 + tid] = 0;                  // zero pad cols [1160,1184)
    }
    int fid = feedid[b];
    const float4* fr = (const float4*)(feed_table + (size_t)fid * FD);
    if (tid < 128) {                           // 128 float4 = 512 floats
        float4 v = fr[tid];
        uint2 p; p.x = pack2(v.x, v.y); p.y = pack2(v.z, v.w);
        *(uint2*)&row[136 + tid * 4] = p;
    }
}

// ---------------- fused DIN attention (one workgroup per batch element) ----------------
// h = relu([q,h,q-h,q*h] @ W1 + b1) ; score = h @ w2 ; softmax over L ; ui = attn @ hist
// (attn_b2 is a constant shift -> cancels in softmax, omitted)
__global__ __launch_bounds__(256) void attn_kernel(
        const int* __restrict__ hist_ids, const float* __restrict__ feed_table,
        const unsigned int* __restrict__ w1p, const float* __restrict__ b1,
        const float* __restrict__ w2, unsigned short* __restrict__ dnn_in) {
    __shared__ __align__(16) unsigned short s_hist[HLEN * FD];   // 50 KB bf16
    __shared__ __align__(16) unsigned short s_q[FD];             // 1 KB
    __shared__ float s_score[64];
    __shared__ float s_attn[64];
    __shared__ int   s_hid[HLEN];

    int b = blockIdx.x, tid = threadIdx.x;
    int lane = tid & 31, w = tid >> 5;

    if (tid < HLEN) s_hid[tid] = hist_ids[b * HLEN + tid];
    if (tid < 64)   s_score[tid] = 0.f;
    {   // q already converted to bf16 by gather_kernel: 1KB copy as uint2
        const unsigned short* qrow = dnn_in + (size_t)b * IN_PAD + 136;
        if (tid < 128) ((uint2*)s_q)[tid] = ((const uint2*)qrow)[tid];
    }
    __syncthreads();
    for (int idx = tid; idx < HLEN * 128; idx += 256) {     // float4-vectorized stage
        int l = idx >> 7, c4 = idx & 127;
        const float4* hr = (const float4*)(feed_table + (size_t)s_hid[l] * FD);
        __builtin_prefetch(hr + c4 + 64, 0, 1);
        float4 v = hr[c4];
        uint2 p; p.x = pack2(v.x, v.y); p.y = pack2(v.z, v.w);
        *(uint2*)&s_hist[l * FD + c4 * 4] = p;
    }
    __syncthreads();

    // GEMM: M = 64 (50 rows + pad), N = 128, K = 2048 (4 quarters of 512)
    int mt = w & 3, ng = w >> 2;                 // wave -> (m-tile, n-group of 4 tiles)
    int row_m = mt * 16 + (lane & 15);
    int row_c = (row_m < HLEN) ? row_m : (HLEN - 1);  // pad rows: clamp (results unused)
    int half  = (lane >= 16) ? 8 : 0;
    const unsigned short* qbase = s_q + half;
    const unsigned short* hbase = s_hist + row_c * FD + half;
    const unsigned int*   wl    = w1p + lane * 8;

    v8f acc[4];
    v8f zero = {0.f, 0.f, 0.f, 0.f, 0.f, 0.f, 0.f, 0.f};
#pragma unroll
    for (int i = 0; i < 4; ++i) acc[i] = zero;

#define MMA4(kc)                                                            \
    {                                                                       \
        _Pragma("unroll")                                                   \
        for (int i = 0; i < 4; ++i) {                                       \
            Frag bfr;                                                       \
            load_bfrag(bfr, wl + (size_t)((ng * 4 + i) * 64 + (kc)) * 256); \
            acc[i] = wmma_bf16(a, bfr, acc[i]);                             \
        }                                                                   \
    }

    for (int kc16 = 0; kc16 < 16; ++kc16) {      // quarter 0: q (row-invariant)
        Frag a; load_frag16(a, qbase + kc16 * 32);
        MMA4(kc16);
    }
    for (int kc16 = 0; kc16 < 16; ++kc16) {      // quarter 1: hist
        Frag a; load_frag16(a, hbase + kc16 * 32);
        MMA4(16 + kc16);
    }
    for (int kc16 = 0; kc16 < 16; ++kc16) {      // quarter 2: q - hist
        Frag aq, ah, a;
        load_frag16(aq, qbase + kc16 * 32);
        load_frag16(ah, hbase + kc16 * 32);
        a.v = aq.v - ah.v;
        MMA4(32 + kc16);
    }
    for (int kc16 = 0; kc16 < 16; ++kc16) {      // quarter 3: q * hist
        Frag aq, ah, a;
        load_frag16(aq, qbase + kc16 * 32);
        load_frag16(ah, hbase + kc16 * 32);
        a.v = aq.v * ah.v;
        MMA4(48 + kc16);
    }
#undef MMA4

    // bias + relu + dot with w2, reduce over N (lanes within 16-groups), LDS atomic over waves
#pragma unroll
    for (int r = 0; r < 8; ++r) {
        float s = 0.f;
#pragma unroll
        for (int i = 0; i < 4; ++i) {
            int n = ng * 64 + i * 16 + (lane & 15);
            float h = fmaxf(acc[i][r] + b1[n], 0.f);
            s += h * w2[n];
        }
        for (int off = 8; off > 0; off >>= 1) s += __shfl_xor(s, off, 16);
        if ((lane & 15) == 0) {
            int rowg = mt * 16 + r + ((lane >= 16) ? 8 : 0);
            if (rowg < HLEN) atomicAdd(&s_score[rowg], s);   // ds_add_f32
        }
    }
    __syncthreads();
    if (tid == 0) {                              // softmax over L=50
        float m = -1e30f;
        for (int l = 0; l < HLEN; ++l) m = fmaxf(m, s_score[l]);
        float sum = 0.f;
        for (int l = 0; l < HLEN; ++l) { float e = __expf(s_score[l] - m); s_attn[l] = e; sum += e; }
        float inv = 1.f / sum;
        for (int l = 0; l < HLEN; ++l) s_attn[l] *= inv;
    }
    __syncthreads();
    // user_interest = attn @ hist  -> write into dnn_in cols [648,1160), pairwise
    unsigned short* urow = dnn_in + (size_t)b * IN_PAD + 648;
    {
        int d2 = tid;                            // 256 threads, 256 bf16 pairs
        float s0 = 0.f, s1 = 0.f;
        for (int l = 0; l < HLEN; ++l) {
            unsigned int p = *(const unsigned int*)&s_hist[l * FD + d2 * 2];
            float aw = s_attn[l];
            s0 += aw * lo16f(p);
            s1 += aw * hi16f(p);
        }
        *(unsigned int*)&urow[d2 * 2] = pack2(s0, s1);
    }
}

// ---------------- generic WMMA GEMM: bf16 A [M,LDA] x packed bf16 B -> bias/relu ----------------
template <int KCH, int NT_PER_WAVE, int LDA>
__global__ __launch_bounds__(256) void gemm_bf16_kernel(
        const unsigned short* __restrict__ A, const unsigned int* __restrict__ Bp,
        const float* __restrict__ bias, void* __restrict__ out,
        int N, int do_relu, int out_bf16) {
    int tid = threadIdx.x, lane = tid & 31, w = tid >> 5;
    int mt = w & 3, ng = w >> 2;
    int mbase = blockIdx.x * 64 + mt * 16;

    v8f acc[NT_PER_WAVE];
    v8f zero = {0.f, 0.f, 0.f, 0.f, 0.f, 0.f, 0.f, 0.f};
#pragma unroll
    for (int i = 0; i < NT_PER_WAVE; ++i) acc[i] = zero;

    int arow = mbase + (lane & 15);
    const unsigned short* Ar = A + (size_t)arow * LDA + ((lane >= 16) ? 8 : 0);
    const unsigned int*   wl = Bp + lane * 8;

    for (int kc = 0; kc < KCH; ++kc) {
        if (kc + 1 < KCH) __builtin_prefetch(Ar + (kc + 1) * 32, 0, 0);
        Frag a;
        load_frag16(a, Ar + kc * 32);            // two global_load_b128
#pragma unroll
        for (int i = 0; i < NT_PER_WAVE; ++i) {
            Frag bfr;
            load_bfrag(bfr, wl + (size_t)((ng * NT_PER_WAVE + i) * KCH + kc) * 256);
            acc[i] = wmma_bf16(a, bfr, acc[i]);
        }
    }
#pragma unroll
    for (int i = 0; i < NT_PER_WAVE; ++i) {
        int n = (ng * NT_PER_WAVE + i) * 16 + (lane & 15);
#pragma unroll
        for (int r = 0; r < 8; ++r) {
            int m = mbase + r + ((lane >= 16) ? 8 : 0);
            float v = acc[i][r];
            if (bias)    v += bias[n];
            if (do_relu) v = fmaxf(v, 0.f);
            if (out_bf16) ((unsigned short*)out)[(size_t)m * N + n] = f2bf(v);
            else          ((float*)out)[(size_t)m * N + n] = v;
        }
    }
}

// ---------------- MMOE tail: gate softmax, task mix, per-task head, sigmoid ----------------
__global__ void final_kernel(const float* __restrict__ eg, const float* __restrict__ out_w,
                             const float* __restrict__ out_b, float* __restrict__ out) {
    int b = blockIdx.x * blockDim.x + threadIdx.x;
    if (b >= BATCH) return;
    const float* row = eg + (size_t)b * 96;      // [0,64)=expert_out(e,o), [64,96)=gate logits(t,e)
    for (int t = 0; t < NTASK; ++t) {
        float g[NEXP]; float m = -1e30f;
        for (int e = 0; e < NEXP; ++e) { g[e] = row[64 + t * NEXP + e]; m = fmaxf(m, g[e]); }
        float sum = 0.f;
        for (int e = 0; e < NEXP; ++e) { g[e] = __expf(g[e] - m); sum += g[e]; }
        float inv = 1.f / sum;
        float logit = out_b[t];
        for (int o = 0; o < EOUT; ++o) {
            float to = 0.f;
            for (int e = 0; e < NEXP; ++e) to += g[e] * inv * row[e * EOUT + o];
            logit += to * out_w[t * EOUT + o];
        }
        out[b * NTASK + t] = 1.f / (1.f + __expf(-logit));
    }
}

// ---------------- launch ----------------
extern "C" void kernel_launch(void* const* d_in, const int* in_sizes, int n_in,
                              void* d_out, int out_size, void* d_ws, size_t ws_size,
                              hipStream_t stream) {
    (void)in_sizes; (void)n_in; (void)out_size; (void)ws_size;
    const int*   sparse_ids = (const int*)  d_in[0];
    const float* dense      = (const float*)d_in[1];
    const int*   feedid     = (const int*)  d_in[2];
    const int*   hist_ids   = (const int*)  d_in[3];
    const float* emb_tables = (const float*)d_in[4];
    const float* feed_table = (const float*)d_in[5];
    const float* attn_w1    = (const float*)d_in[6];
    const float* attn_b1    = (const float*)d_in[7];
    const float* attn_w2    = (const float*)d_in[8];
    /* d_in[9] attn_b2: constant shift, cancels in softmax */
    const float* dnn_w1     = (const float*)d_in[10];
    const float* dnn_b1     = (const float*)d_in[11];
    const float* dnn_w2     = (const float*)d_in[12];
    const float* dnn_b2     = (const float*)d_in[13];
    const float* expert_w   = (const float*)d_in[14];
    const float* gate_w     = (const float*)d_in[15];
    const float* out_w      = (const float*)d_in[16];
    const float* out_b      = (const float*)d_in[17];

    char*  ws  = (char*)d_ws;
    size_t off = 0;
    auto alloc = [&](size_t bytes) { size_t o = off; off += (bytes + 255) & ~(size_t)255; return o; };
    unsigned int*   w1p     = (unsigned int*)  (ws + alloc((size_t)8  * 64 * 32 * 8 * 4)); // 512KB
    unsigned int*   dw1p    = (unsigned int*)  (ws + alloc((size_t)16 * 37 * 32 * 8 * 4)); // 592KB
    unsigned int*   dw2p    = (unsigned int*)  (ws + alloc((size_t)8  * 8  * 32 * 8 * 4)); //  64KB
    unsigned int*   mwp     = (unsigned int*)  (ws + alloc((size_t)6  * 4  * 32 * 8 * 4)); //  24KB
    float*          mstage  = (float*)         (ws + alloc((size_t)HID2 * 96 * 4));
    unsigned short* dnn_in  = (unsigned short*)(ws + alloc((size_t)BATCH * IN_PAD * 2));   // 4.85MB
    unsigned short* zbuf    = (unsigned short*)(ws + alloc((size_t)BATCH * HID1 * 2));     // 1MB
    unsigned short* dout_bf = (unsigned short*)(ws + alloc((size_t)BATCH * HID2 * 2));     // 512KB
    float*          eg      = (float*)         (ws + alloc((size_t)BATCH * 96 * 4));       // 768KB

    // weight packing (tiny, runs once per launch)
    pack_b_kernel<<<512, 256, 0, stream>>>(attn_w1, w1p, 2048, 128, 2048);
    pack_b_kernel<<<512, 256, 0, stream>>>(dnn_w1, dw1p, IN_PAD, HID1, IN_DIM);
    pack_b_kernel<<<64,  256, 0, stream>>>(dnn_w2, dw2p, HID1, HID2, HID1);
    mmoe_stage_kernel<<<48, 256, 0, stream>>>(expert_w, gate_w, mstage);
    pack_b_kernel<<<24,  256, 0, stream>>>(mstage, mwp, HID2, 96, HID2);

    // embedding gathers -> bf16 staging row
    gather_kernel<<<BATCH, 256, 0, stream>>>(sparse_ids, dense, feedid,
                                             emb_tables, feed_table, dnn_in);
    // fused DIN attention (big WMMA GEMM + softmax + weighted sum)
    attn_kernel<<<BATCH, 256, 0, stream>>>(hist_ids, feed_table, w1p,
                                           attn_b1, attn_w2, dnn_in);
    // DNN layer 1: [2048,1184] x [1184,256] + b, relu
    gemm_bf16_kernel<37, 8, IN_PAD><<<BATCH / 64, 256, 0, stream>>>(
        dnn_in, dw1p, dnn_b1, zbuf, HID1, 1, 1);
    // DNN layer 2: [2048,256] x [256,128] + b, relu
    gemm_bf16_kernel<8, 4, HID1><<<BATCH / 64, 256, 0, stream>>>(
        zbuf, dw2p, dnn_b2, dout_bf, HID2, 1, 1);
    // MMOE expert+gate matmul: [2048,128] x [128,96] -> f32
    gemm_bf16_kernel<4, 3, HID2><<<BATCH / 64, 256, 0, stream>>>(
        dout_bf, mwp, nullptr, eg, 96, 0, 0);
    // gates softmax + task mixing + heads + sigmoid
    final_kernel<<<BATCH / 256, 256, 0, stream>>>(eg, out_w, out_b, (float*)d_out);
}
